// Model_85916525789512
// MI455X (gfx1250) — matrix operand order repaired and checked by reading, not compiled, not run on hardware
//
// CDNA5 (gfx1250) fused actor-critic forward.
// WMMA f32<-f16 GEMM pipeline + Tensor Data Mover weight staging into LDS.
#include <hip/hip_runtime.h>
#include <cstdint>

typedef __attribute__((ext_vector_type(16))) _Float16     v16h;
typedef __attribute__((ext_vector_type(8)))  float        v8f;
typedef __attribute__((ext_vector_type(4)))  unsigned int u32x4;
typedef __attribute__((ext_vector_type(8)))  int          i32x8;
typedef __attribute__((ext_vector_type(4)))  int          i32x4;

#define WMMA_F16(a, b, c) \
  __builtin_amdgcn_wmma_f32_16x16x32_f16(false, (a), false, (b), (short)0, (c), false, false)

#if __has_builtin(__builtin_amdgcn_tensor_load_to_lds) && \
    __has_builtin(__builtin_amdgcn_s_wait_tensorcnt)
#define USE_TDM 1
#else
#define USE_TDM 0
#endif

constexpr int S_ = 256, B_ = 1024, D_ = 64, H_ = 128, A_ = 8;
constexpr long NTOK = (long)S_ * B_;  // 262144 tokens
constexpr int LDSW  = 136;            // padded LDS row stride (halves) for 128-wide rows
constexpr int LDSW1 = 72;             // padded LDS row stride (halves) for 64-wide rows

// f16 weight workspace layout in d_ws (offsets in halves)
constexpr int OFF_AW1  = 0;
constexpr int OFF_AW2  = OFF_AW1  + H_ * D_;
constexpr int OFF_CW1  = OFF_AW2  + H_ * H_;
constexpr int OFF_CW2  = OFF_CW1  + H_ * D_;
constexpr int OFF_AWIH = OFF_CW2  + H_ * H_;
constexpr int OFF_CWIH = OFF_AWIH + 4 * H_ * H_;
constexpr int OFF_MW   = OFF_CWIH + 4 * H_ * H_;
constexpr int OFF_LW   = OFF_MW   + A_ * H_;   // end: 182272 halves = 364544 B

// dynamic LDS layout (offsets in halves)
constexpr int ACT_H = 8 * 16 * LDSW;          // per-wave activation tiles   (34816 B)
constexpr int S1_H  = ACT_H;                  // W1 staging: 128 x 72        (18432 B)
constexpr int S2_H  = S1_H + 128 * LDSW1;     // W2 staging: 128 x 136       (34816 B)
constexpr int S3_H  = S2_H + 128 * LDSW;      // WIH i/g/o:  384 x 136      (104448 B)
constexpr int TOT_H = S3_H + 384 * LDSW;      // total 96256 halves = 192512 B

// ---------------- weight repack fp32 -> f16 ----------------
__global__ void cvt_f32_to_f16(const float* __restrict__ src,
                               _Float16* __restrict__ dst, int n) {
  int i = blockIdx.x * blockDim.x + threadIdx.x;
  if (i < n) dst[i] = (_Float16)src[i];
}

// ---------------- fragment loaders ----------------
// B fragment of W^T (16 cols x 32 K) from LDS-staged weights, row stride `stride`.
// Layout per ISA: N = lane&15; elems 0..7: K=(lane>>4)*8+e; elems 8..15: K=16+(lane>>4)*8+(e-8).
__device__ __forceinline__ v16h load_bfrag_lds(const _Float16* Wl, int stride,
                                               int n0, int k0, int lane) {
  int n = n0 + (lane & 15);
  int g = lane >> 4;
  const uint4* p = (const uint4*)(Wl + n * stride + k0 + g * 8);
  union { uint4 q[2]; v16h h; } u;
  u.q[0] = p[0];
  u.q[1] = p[2];
  return u.h;
}

// B fragment for an 8-row head weight (mw/lw) straight from global; rows 8..15 zero.
__device__ __forceinline__ v16h load_bfrag8(const _Float16* W, int k0, int lane) {
  int n = lane & 15;
  int g = lane >> 4;
  v16h z = {};
  if (n < A_) {
    const uint4* p = (const uint4*)(W + (long)n * H_ + k0 + g * 8);
    union { uint4 q[2]; v16h h; } u;
    u.q[0] = p[0];
    u.q[1] = p[2];
    z = u.h;
  }
  return z;
}

// A fragment (16x32 f16 activations) from wave-private LDS tile [16][LDSW].
// Layout per ISA: M = lane&15; v<4: K=(v&3)*2+h+g*8; v>=4: K=16+(v&3)*2+h+g*8.
__device__ __forceinline__ v16h load_afrag_lds(const _Float16* buf, int k0, int lane) {
  int m = lane & 15;
  int g = lane >> 4;
  const uint4* p = (const uint4*)(buf + m * LDSW + k0 + g * 8);
  union { uint4 q[2]; v16h h; } u;
  u.q[0] = p[0];   // K = k0 + g*8 + 0..7
  u.q[1] = p[2];   // K = k0 + 16 + g*8 + 0..7
  return u.h;
}

__device__ __forceinline__ float sigm(float x) { return 1.f / (1.f + __expf(-x)); }

// ---------------- TDM: 2D weight tile -> padded LDS ----------------
// rows x cols (halves), contiguous in global; TDM pad inserts 4 dwords per row so the
// LDS image has row stride cols/2+4 dwords (72 or 136 halves).
__device__ __forceinline__ void tdm_load_2d(const _Float16* gsrc, unsigned lds_byte,
                                            unsigned rows, unsigned cols) {
#if USE_TDM
  unsigned pad_iv = (cols == 128) ? 5u : 4u;  // pad after 64 / 32 dwords (= one row)
  unsigned long long ga = (unsigned long long)(uintptr_t)gsrc;
  u32x4 g0 = {};
  i32x8 g1 = {};
  i32x4 z4 = {};
  g0[0] = 1u;                                           // count=1 (valid), user mode
  g0[1] = lds_byte;                                     // lds_addr
  g0[2] = (unsigned)(ga & 0xFFFFFFFFu);                 // global_addr[31:0]
  g0[3] = (unsigned)((ga >> 32) & 0x1FFFFFFu)           // global_addr[56:32]
          | (2u << 30);                                 // type = 2 ("image")
  g1[0] = (int)((1u << 16)                              // data_size = 2 bytes
                | (1u << 20)                            // pad_enable
                | (pad_iv << 22)                        // pad_interval
                | (3u << 25));                          // pad_amount = 4 dwords
  g1[1] = (int)((cols & 0xFFFFu) << 16);                // tensor_dim0[15:0]
  g1[2] = (int)((rows & 0xFFFFu) << 16);                // dim0[31:16]=0 | tensor_dim1[15:0]
  g1[3] = (int)((cols & 0xFFFFu) << 16);                // dim1[31:16]=0 | tile_dim0
  g1[4] = (int)(rows & 0xFFFFu);                        // tile_dim1 (tile_dim2 = 0)
  g1[5] = (int)cols;                                    // tensor_dim0_stride[31:0]
  g1[6] = 0;                                            // stride[47:32] | dim1_stride lo
  g1[7] = 0;
#if defined(__clang_major__) && (__clang_major__ >= 23)
  i32x8 z8 = {};
  __builtin_amdgcn_tensor_load_to_lds(g0, g1, z4, z4, z8, 0);
#else
  __builtin_amdgcn_tensor_load_to_lds(g0, g1, z4, z4, 0);
#endif
#else
  (void)gsrc; (void)lds_byte; (void)rows; (void)cols;
#endif
}

// make staged tiles visible: wave0 drains TENSORcnt to CNT, then block barrier.
// (CNT must be a compile-time constant for s_wait_tensorcnt.)
template <int CNT>
__device__ __forceinline__ void stage_sync() {
#if USE_TDM
  if (threadIdx.x < 32) __builtin_amdgcn_s_wait_tensorcnt((short)CNT);
#endif
  __syncthreads();
}

#if !USE_TDM
// fallback: cooperative copy global f16 -> padded LDS (all 256 threads)
__device__ __forceinline__ void coop_stage(const _Float16* gsrc, _Float16* lds,
                                           int rows, int cols, int stride) {
  int chunks = rows * (cols / 8);
  int cpr = cols / 8;
  for (int i = threadIdx.x; i < chunks; i += blockDim.x) {
    int r = i / cpr, c = (i % cpr) * 8;
    *(uint4*)(lds + r * stride + c) = *(const uint4*)(gsrc + (long)r * cols + c);
  }
}
#endif

// ---------------- fused actor-critic forward ----------------
__global__ __launch_bounds__(256)
void fused_ac_kernel(const float* __restrict__ state,
                     const _Float16* __restrict__ wgt,
                     const float* __restrict__ ab1, const float* __restrict__ ab2,
                     const float* __restrict__ cb1, const float* __restrict__ cb2,
                     const float* __restrict__ a_bih, const float* __restrict__ a_bhh,
                     const float* __restrict__ c_bih, const float* __restrict__ c_bhh,
                     const float* __restrict__ mb, const float* __restrict__ lb,
                     const float* __restrict__ vw, const float* __restrict__ vb,
                     float* __restrict__ out) {
  extern __shared__ _Float16 lds[];

  const int wave = threadIdx.x >> 5;
  const int lane = threadIdx.x & 31;
  const int g    = lane >> 4;
  const int nl   = lane & 15;
  _Float16* buf = lds + wave * (16 * LDSW);   // wave-private activation tile
  _Float16* s1  = lds + S1_H;                 // staged W1  [128][72]
  _Float16* s2  = lds + S2_H;                 // staged W2  [128][136]
  _Float16* s3  = lds + S3_H;                 // staged WIH i/g/o [384][136]
  const long t0 = (long)blockIdx.x * 128 + wave * 16;

  // stage all four weight tiles of one branch into LDS
  auto stage_branch = [&](int br) {
    const _Float16* W1  = wgt + (br ? OFF_CW1  : OFF_AW1);
    const _Float16* W2  = wgt + (br ? OFF_CW2  : OFF_AW2);
    const _Float16* WIH = wgt + (br ? OFF_CWIH : OFF_AWIH);
#if USE_TDM
    if (threadIdx.x < 32) {                   // wave 0 drives the TDM
      tdm_load_2d(W1,              (unsigned)(S1_H * 2), 128, (unsigned)D_);
      tdm_load_2d(W2,              (unsigned)(S2_H * 2), 128, (unsigned)H_);
      tdm_load_2d(WIH,             (unsigned)(S3_H * 2), 128, (unsigned)H_);           // i
      tdm_load_2d(WIH + 256 * H_,  (unsigned)((S3_H + 128 * LDSW) * 2), 256,
                  (unsigned)H_);                                                       // g,o
    }
#else
    coop_stage(W1, s1, 128, D_, LDSW1);
    coop_stage(W2, s2, 128, H_, LDSW);
    coop_stage(WIH, s3, 128, H_, LDSW);
    coop_stage(WIH + 256 * H_, s3 + 128 * LDSW, 256, H_, LDSW);
#endif
  };

  // --- state tile [16 x 64] as two 16x32 f16 A-fragments (shared by both branches) ---
  v16h xa[2];
  {
    const float* row = state + (t0 + nl) * D_;
#pragma unroll
    for (int c = 0; c < 2; ++c) {
      const float* p = row + c * 32 + g * 8;
      v16h h;
#pragma unroll
      for (int i = 0; i < 8; ++i) h[i] = (_Float16)p[i];
#pragma unroll
      for (int i = 0; i < 8; ++i) h[8 + i] = (_Float16)p[16 + i];
      xa[c] = h;
    }
  }

  float* out_mean = out;
  float* out_std  = out + NTOK * A_;
  float* out_val  = out + NTOK * A_ * 2;

  stage_branch(0);  // kick off actor weight DMA immediately

  for (int br = 0; br < 2; ++br) {
    const float* b1  = br ? cb1   : ab1;
    const float* b2  = br ? cb2   : ab2;
    const float* bih = br ? c_bih : a_bih;
    const float* bhh = br ? c_bhh : a_bhh;

    // ---- stage 1: h1 = relu(X @ W1^T + b1) -> act LDS ----
    stage_sync<3>();                    // W1 resident
    for (int j = 0; j < 8; ++j) {
      v8f acc = {};
#pragma unroll
      for (int c = 0; c < 2; ++c) {
        v16h b = load_bfrag_lds(s1, LDSW1, j * 16, c * 32, lane);
        acc = WMMA_F16(xa[c], b, acc);
      }
      float bias = b1[j * 16 + nl];
#pragma unroll
      for (int r = 0; r < 8; ++r) {
        float v = acc[r] + bias;
        v = v > 0.f ? v : 0.f;
        buf[(r + 8 * g) * LDSW + j * 16 + nl] = (_Float16)v;
      }
    }

    // ---- stage 2: h2 = relu(h1 @ W2^T + b2) ----
    stage_sync<2>();                    // W2 resident
    v16h a2[4];
#pragma unroll
    for (int c = 0; c < 4; ++c) a2[c] = load_afrag_lds(buf, c * 32, lane);
    for (int j = 0; j < 8; ++j) {
      v8f acc = {};
#pragma unroll
      for (int c = 0; c < 4; ++c) {
        v16h b = load_bfrag_lds(s2, LDSW, j * 16, c * 32, lane);
        acc = WMMA_F16(a2[c], b, acc);
      }
      float bias = b2[j * 16 + nl];
#pragma unroll
      for (int r = 0; r < 8; ++r) {
        float v = acc[r] + bias;
        v = v > 0.f ? v : 0.f;
        buf[(r + 8 * g) * LDSW + j * 16 + nl] = (_Float16)v;
      }
    }

    // ---- stage 3: zero-state LSTM, gates i/g/o only (f dead: c_prev = 0) ----
    stage_sync<0>();                    // all WIH tiles resident
    v16h a3[4];
#pragma unroll
    for (int c = 0; c < 4; ++c) a3[c] = load_afrag_lds(buf, c * 32, lane);
    for (int j = 0; j < 8; ++j) {
      v8f ai = {}, ag = {}, ao = {};
#pragma unroll
      for (int c = 0; c < 4; ++c) {
        v16h bi = load_bfrag_lds(s3, LDSW, 0 * H_ + j * 16, c * 32, lane);  // i
        ai = WMMA_F16(a3[c], bi, ai);
        v16h bg = load_bfrag_lds(s3, LDSW, 1 * H_ + j * 16, c * 32, lane);  // g
        ag = WMMA_F16(a3[c], bg, ag);
        v16h bo = load_bfrag_lds(s3, LDSW, 2 * H_ + j * 16, c * 32, lane);  // o
        ao = WMMA_F16(a3[c], bo, ao);
      }
      int col = j * 16 + nl;
      float bi_ = bih[0 * H_ + col] + bhh[0 * H_ + col];
      float bg_ = bih[2 * H_ + col] + bhh[2 * H_ + col];
      float bo_ = bih[3 * H_ + col] + bhh[3 * H_ + col];
#pragma unroll
      for (int r = 0; r < 8; ++r) {
        float iv = sigm(ai[r] + bi_);
        float gv = tanhf(ag[r] + bg_);
        float ov = sigm(ao[r] + bo_);
        float hv = ov * tanhf(iv * gv);
        buf[(r + 8 * g) * LDSW + col] = (_Float16)hv;
      }
    }

    // ---- heads ----
    if (br == 0) {
      v16h ah[4];
#pragma unroll
      for (int c = 0; c < 4; ++c) ah[c] = load_afrag_lds(buf, c * 32, lane);
      v8f am = {}, as = {};
#pragma unroll
      for (int c = 0; c < 4; ++c) {
        v16h bm = load_bfrag8(wgt + OFF_MW, c * 32, lane);
        am = WMMA_F16(ah[c], bm, am);
        v16h bl = load_bfrag8(wgt + OFF_LW, c * 32, lane);
        as = WMMA_F16(ah[c], bl, as);
      }
      if (nl < A_) {
        float bm_ = mb[nl], bl_ = lb[nl];
#pragma unroll
        for (int r = 0; r < 8; ++r) {
          long t = t0 + r + 8 * g;
          out_mean[t * A_ + nl] = am[r] + bm_;
          out_std [t * A_ + nl] = expf(as[r] + bl_);
        }
      }
      __syncthreads();     // all waves done reading actor staging
      stage_branch(1);     // launch critic weight DMA
    } else {
      if (lane < 16) {
        float s = vb[0];
#pragma unroll 8
        for (int k = 0; k < H_; ++k)
          s += (float)buf[lane * LDSW + k] * vw[k];
        out_val[t0 + lane] = s;
      }
    }
  }
}

// ---------------- launch ----------------
extern "C" void kernel_launch(void* const* d_in, const int* in_sizes, int n_in,
                              void* d_out, int out_size, void* d_ws, size_t ws_size,
                              hipStream_t stream) {
  const float* state = (const float*)d_in[0];
  _Float16* wgt = (_Float16*)d_ws;

  auto cvt = [&](int idx, int off, int n) {
    cvt_f32_to_f16<<<(n + 255) / 256, 256, 0, stream>>>(
        (const float*)d_in[idx], wgt + off, n);
  };
  cvt(1,  OFF_AW1,  H_ * D_);      // aw1
  cvt(3,  OFF_AW2,  H_ * H_);      // aw2
  cvt(5,  OFF_CW1,  H_ * D_);      // cw1
  cvt(7,  OFF_CW2,  H_ * H_);      // cw2
  cvt(9,  OFF_AWIH, 4 * H_ * H_);  // a_wih
  cvt(13, OFF_CWIH, 4 * H_ * H_);  // c_wih
  cvt(17, OFF_MW,   A_ * H_);      // mw
  cvt(19, OFF_LW,   A_ * H_);      // lw

  dim3 grid((unsigned)(NTOK / 128));
  fused_ac_kernel<<<grid, 256, (unsigned)(TOT_H * sizeof(_Float16)), stream>>>(
      state, wgt,
      (const float*)d_in[2],  (const float*)d_in[4],   // ab1, ab2
      (const float*)d_in[6],  (const float*)d_in[8],   // cb1, cb2
      (const float*)d_in[11], (const float*)d_in[12],  // a_bih, a_bhh
      (const float*)d_in[15], (const float*)d_in[16],  // c_bih, c_bhh
      (const float*)d_in[18], (const float*)d_in[20],  // mb, lb
      (const float*)d_in[21], (const float*)d_in[22],  // vw, vb
      (float*)d_out);
}
